// GCNNet_86595130622529
// MI455X (gfx1250) — matrix operand order repaired and checked by reading, not compiled, not run on hardware
//
#include <hip/hip_runtime.h>
#include <hip/hip_bf16.h>

// ---------------------------------------------------------------------------
// GCNNet forward for MI455X (gfx1250), wave32 + bf16 WMMA.
// GEMM: 128x128 block tile, 8 waves x (32x64) wave tiles = 8 WMMA/K-step,
// double-buffered LDS, single barrier per step; A staged via async
// global->LDS b128 (ASYNCcnt path), B staged as k-pair-packed dwords.
// All K dims padded to multiples of 16 and all activation leading dims to
// multiples of 8 (zero-filled pads) so every large GEMM runs the fast paths.
// Graph aggregation / pooling / batchnorm are fp32 elementwise/atomic kernels.
// ---------------------------------------------------------------------------

typedef __bf16 bf16;
typedef __attribute__((ext_vector_type(16))) __bf16 v16bf;
typedef __attribute__((ext_vector_type(8)))  float  v8f;
typedef __attribute__((ext_vector_type(4)))  unsigned int v4u;

#define GNODES 200000
#define GEDGES 800000
#define GBATCH 4096
#define NF0    78
#define NCCATP 5480

// ---------------- GEMM: C = act(A(bf16,MxK) @ B(bf16,KxN) + bias) ----------
#define BM 128
#define BN 128
#define BK 32
#define LDA_SH (BK + 8)     // bf16 elems; row stride 80 B (16B aligned)
#define LDB_P  (BN + 4)     // packed dwords per k-pair row

__device__ __forceinline__ void async_b128(unsigned lds_off, const void* g)
{
    unsigned long long ga = (unsigned long long)(uintptr_t)g;
    asm volatile("global_load_async_to_lds_b128 %0, %1, off"
                 :: "v"(lds_off), "v"(ga) : "memory");
}

__device__ __forceinline__ void wait_async0()
{
#if __has_builtin(__builtin_amdgcn_s_wait_asynccnt)
    __builtin_amdgcn_s_wait_asynccnt(0);
#else
    asm volatile("s_wait_asynccnt 0x0" ::: "memory");
#endif
}

__device__ __forceinline__ unsigned short b2u(bf16 x)
{
    union { bf16 b; unsigned short u; } c; c.b = x; return c.u;
}

__device__ __forceinline__ bf16 ldclamp(const bf16* P, int r, int c,
                                        int ld, int R, int C)
{
    int rr = r < R ? r : R - 1;
    int cc = c < C ? c : C - 1;
    bf16 v = P[(size_t)rr * ld + cc];
    return (r < R && c < C) ? v : (bf16)0.0f;
}

__global__ __launch_bounds__(256)
void gemm_bf16_kernel(const bf16* __restrict__ A, int lda,
                      const bf16* __restrict__ Bm, int ldb,
                      const float* __restrict__ bias,
                      float* __restrict__ Cf, bf16* __restrict__ Cb, int ldc,
                      int M, int N, int K, int relu, int nstore)
{
    __shared__ __align__(16) bf16     sA[2][BM * LDA_SH];
    __shared__ __align__(16) unsigned sBp[2][(BK / 2) * LDB_P];

    const int tid  = threadIdx.x;
    const int wave = tid >> 5;       // 0..7
    const int lane = tid & 31;
    const int wm   = wave >> 1;      // 0..3 -> 32-row slab
    const int wn   = wave & 1;       // 0..1 -> 64-col slab
    const int half = lane >> 4;
    const int r16  = lane & 15;

    const int blockM = blockIdx.x * BM;
    const int blockN = blockIdx.y * BN;

    v8f acc[2][4];
    const v8f vzero = {0.f, 0.f, 0.f, 0.f, 0.f, 0.f, 0.f, 0.f};
#pragma unroll
    for (int mi = 0; mi < 2; ++mi)
#pragma unroll
        for (int ni = 0; ni < 4; ++ni) acc[mi][ni] = vzero;

    const bool aFast128 = ((lda & 7) == 0) && (blockM + BM <= M);
    const bool aFast32  = ((lda & 1) == 0) && (blockM + BM <= M);
    const bool bFast32  = ((ldb & 1) == 0);   // pad cols only feed gn>=N

    auto stageA = [&](int k0, bf16* sa) {
        const bool kfull = (k0 + BK) <= K;
        if (kfull && aFast128) {
            // async global->LDS, two b128 per thread (128 rows x 4 chunks)
#pragma unroll
            for (int i = 0; i < 2; ++i) {
                int idx = tid + i * 256;
                int row = idx >> 2, c8 = (idx & 3) * 8;
                unsigned loff = (unsigned)(uintptr_t)(void*)&sa[row * LDA_SH + c8];
                async_b128(loff, A + (size_t)(blockM + row) * lda + k0 + c8);
            }
        } else if (kfull && aFast32) {
            unsigned tmp[8];                    // gather, then scatter to LDS
#pragma unroll
            for (int i = 0; i < 8; ++i) {
                int idx = tid + i * 256;
                int row = idx >> 4, cd = idx & 15;
                tmp[i] = *(const unsigned*)(A + (size_t)(blockM + row) * lda + k0 + cd * 2);
            }
#pragma unroll
            for (int i = 0; i < 8; ++i) {
                int idx = tid + i * 256;
                int row = idx >> 4, cd = idx & 15;
                *(unsigned*)&sa[row * LDA_SH + cd * 2] = tmp[i];
            }
        } else {
            // branch-free clamped tail: thread owns 16 contiguous elems
            int row = tid >> 1, kk = (tid & 1) * 16;
            union { bf16 e[16]; v4u q[4]; } t;
#pragma unroll
            for (int i = 0; i < 16; ++i)
                t.e[i] = ldclamp(A, blockM + row, k0 + kk + i, lda, M, K);
            v4u* dst = (v4u*)&sa[row * LDA_SH + kk];
#pragma unroll
            for (int i = 0; i < 4; ++i) dst[i] = t.q[i];
        }
    };

    auto stageB = [&](int k0, unsigned* sb) {
        const bool kfull = (k0 + BK) <= K;
        if (kfull && bFast32) {
#pragma unroll
            for (int i = 0; i < 4; ++i) {
                int idx = tid + i * 256;         // 1024 units (k2, col-pair)
                int k2 = idx >> 6, cp = idx & 63;
                const bf16* r0 = Bm + (size_t)(k0 + 2 * k2) * ldb + blockN + cp * 2;
                unsigned ua = *(const unsigned*)r0;
                unsigned ub = *(const unsigned*)(r0 + ldb);
                sb[k2 * LDB_P + cp * 2]     = (ua & 0xffffu) | (ub << 16);
                sb[k2 * LDB_P + cp * 2 + 1] = (ua >> 16) | (ub & 0xffff0000u);
            }
        } else {
            unsigned p0[4], p1[4];               // gather, then scatter to LDS
#pragma unroll
            for (int i = 0; i < 4; ++i) {
                int idx = tid + i * 256;
                int k2 = idx >> 6, cp = idx & 63;
                int r0 = k0 + 2 * k2, n0 = blockN + cp * 2;
                bf16 e00 = ldclamp(Bm, r0,     n0,     ldb, K, N);
                bf16 e01 = ldclamp(Bm, r0,     n0 + 1, ldb, K, N);
                bf16 e10 = ldclamp(Bm, r0 + 1, n0,     ldb, K, N);
                bf16 e11 = ldclamp(Bm, r0 + 1, n0 + 1, ldb, K, N);
                p0[i] = (unsigned)b2u(e00) | ((unsigned)b2u(e10) << 16);
                p1[i] = (unsigned)b2u(e01) | ((unsigned)b2u(e11) << 16);
            }
#pragma unroll
            for (int i = 0; i < 4; ++i) {
                int idx = tid + i * 256;
                int k2 = idx >> 6, cp = idx & 63;
                sb[k2 * LDB_P + cp * 2]     = p0[i];
                sb[k2 * LDB_P + cp * 2 + 1] = p1[i];
            }
        }
    };

    // prologue: stage first tile into buffer 0
    int buf = 0;
    stageA(0, sA[0]);
    stageB(0, sBp[0]);
    wait_async0();
    __syncthreads();

    for (int k0 = 0; k0 < K; k0 += BK) {
        int kn = k0 + BK;
        if (kn < K) {                 // prefetch next tile into other buffer
            stageA(kn, sA[buf ^ 1]);
            stageB(kn, sBp[buf ^ 1]);
        }

        // ---- fragments from current buffer ----
        // A: lane=row(mod16); halves hold K 0-15 vs 16-31 -> 2xb128
        union { v16bf v; v4u q[2]; } af[2];
#pragma unroll
        for (int mi = 0; mi < 2; ++mi) {
            const bf16* ap = &sA[buf][(wm * 32 + mi * 16 + r16) * LDA_SH];
            const int kb0 = half ? 16 : 0;
            const int kb1 = half ? 24 : 8;
            af[mi].q[0] = *(const v4u*)&ap[kb0];
            af[mi].q[1] = *(const v4u*)&ap[kb1];
        }
        // B: lane=col(mod16); halves hold K 0-15 vs 16-31 -> 8x ds_load_b32
        union { v16bf v; unsigned u[8]; } bfu[4];
#pragma unroll
        for (int ni = 0; ni < 4; ++ni) {
            const int col = wn * 64 + ni * 16 + r16;
#pragma unroll
            for (int i = 0; i < 8; ++i)
                bfu[ni].u[i] = sBp[buf][(half * 8 + i) * LDB_P + col];
        }
#pragma unroll
        for (int mi = 0; mi < 2; ++mi)
#pragma unroll
            for (int ni = 0; ni < 4; ++ni)
                acc[mi][ni] = __builtin_amdgcn_wmma_f32_16x16x32_bf16(
                    false, af[mi].v, false, bfu[ni].v, (short)0, acc[mi][ni],
                    false, false);

        wait_async0();                // drain prefetch before barrier
        __syncthreads();              // single barrier per K-step
        buf ^= 1;
    }

    // C layout: n = lane&15, m = half*8 + vgpr_index
#pragma unroll
    for (int mi = 0; mi < 2; ++mi) {
#pragma unroll
        for (int ni = 0; ni < 4; ++ni) {
            int gn = blockN + wn * 64 + ni * 16 + r16;
            if (gn >= nstore) continue;
            bool inN = gn < N;
            float bv = (inN && bias) ? bias[gn] : 0.0f;
#pragma unroll
            for (int i = 0; i < 8; ++i) {
                int gm = blockM + wm * 32 + mi * 16 + half * 8 + i;
                if (gm >= M) continue;
                float v = 0.0f;
                if (inN) {
                    v = acc[mi][ni][i] + bv;
                    if (relu) v = v > 0.0f ? v : 0.0f;
                }
                if (Cf) Cf[(size_t)gm * ldc + gn] = v;
                if (Cb) Cb[(size_t)gm * ldc + gn] = (bf16)v;
            }
        }
    }
}

// ---------------- elementwise / graph kernels ------------------------------
// rows x C fp32 -> (rowsPad x C) bf16, zero rows beyond rk
__global__ void cvt_pad_rows_kernel(const float* __restrict__ in,
                                    bf16* __restrict__ out,
                                    int rk, int C, int rkpad)
{
    size_t total = (size_t)rkpad * C;
    size_t i = (size_t)blockIdx.x * blockDim.x + threadIdx.x;
    size_t st = (size_t)gridDim.x * blockDim.x;
    for (; i < total; i += st) {
        int r = (int)(i / C), c = (int)(i % C);
        out[i] = (r < rk) ? (bf16)in[(size_t)r * C + c] : (bf16)0.0f;
    }
}

// [Nn,F] fp32 -> [Nn,Fpad] bf16, zero pad columns
__global__ void cvt_pad_cols_kernel(const float* __restrict__ in,
                                    bf16* __restrict__ out,
                                    int Nn, int F, int Fpad)
{
    size_t total = (size_t)Nn * Fpad;
    size_t i = (size_t)blockIdx.x * blockDim.x + threadIdx.x;
    size_t st = (size_t)gridDim.x * blockDim.x;
    for (; i < total; i += st) {
        int n = (int)(i / Fpad), f = (int)(i % Fpad);
        out[i] = (f < F) ? (bf16)in[(size_t)n * F + f] : (bf16)0.0f;
    }
}

__global__ void zero_f32_kernel(float* __restrict__ p, size_t n)
{
    size_t i = (size_t)blockIdx.x * blockDim.x + threadIdx.x;
    size_t st = (size_t)gridDim.x * blockDim.x;
    for (; i < n; i += st) p[i] = 0.0f;
}

__global__ void degree_kernel(const int* __restrict__ dst,
                              float* __restrict__ deg, int E)
{
    int e = blockIdx.x * blockDim.x + threadIdx.x;
    if (e < E) atomicAdd(&deg[dst[e]], 1.0f);
}

__global__ void dis_kernel(float* __restrict__ deg, int n)
{
    int i = blockIdx.x * blockDim.x + threadIdx.x;
    if (i < n) deg[i] = rsqrtf(deg[i] + 1.0f);   // self-loop degree
}

// out[n,f] = h[n,f] * dis[n]^2   (self-loop term initializes accumulator)
__global__ void selfloop_kernel(const bf16* __restrict__ h,
                                const float* __restrict__ dis,
                                float* __restrict__ out, int Nn, int F)
{
    size_t total = (size_t)Nn * F;
    size_t i = (size_t)blockIdx.x * blockDim.x + threadIdx.x;
    size_t st = (size_t)gridDim.x * blockDim.x;
    for (; i < total; i += st) {
        int n = (int)(i / F);
        float d = dis[n];
        out[i] = (float)h[i] * d * d;
    }
}

// out[dst] += h[src] * dis[src]*dis[dst]; one block per edge, threads over F
__global__ void scatter_kernel(const bf16* __restrict__ h,
                               const int* __restrict__ src,
                               const int* __restrict__ dst,
                               const float* __restrict__ dis,
                               float* __restrict__ out, int E, int F)
{
    int e = blockIdx.x;
    if (e >= E) return;
    int s = src[e], d = dst[e];
    float norm = dis[s] * dis[d];
    const bf16* hp = h + (size_t)s * F;
    float* op = out + (size_t)d * F;
    for (int f = threadIdx.x; f < F; f += blockDim.x)
        atomicAdd(&op[f], (float)hp[f] * norm);
}

// x[n, 0..Fpad) = f<F ? relu(y[n,f]+b[f]) : 0   (padded bf16 activation)
__global__ void bias_relu_pad_kernel(const float* __restrict__ y,
                                     const float* __restrict__ b,
                                     bf16* __restrict__ x,
                                     int Nn, int F, int Fpad)
{
    size_t total = (size_t)Nn * Fpad;
    size_t i = (size_t)blockIdx.x * blockDim.x + threadIdx.x;
    size_t st = (size_t)gridDim.x * blockDim.x;
    for (; i < total; i += st) {
        int n = (int)(i / Fpad), f = (int)(i % Fpad);
        bf16 r = (bf16)0.0f;
        if (f < F) {
            float v = y[(size_t)n * F + f] + b[f];
            r = (bf16)(v > 0.0f ? v : 0.0f);
        }
        x[i] = r;
    }
}

// segment max over contiguous node ranges: graph g owns [ceil(gN/B), ceil((g+1)N/B))
__global__ void pool_max_kernel(const bf16* __restrict__ x,
                                bf16* __restrict__ pooled,
                                int Nn, int F, int ldin, int Fpad, int Bg)
{
    int g = blockIdx.x;
    int start = (int)(((long long)g * Nn + Bg - 1) / Bg);
    int end   = (int)(((long long)(g + 1) * Nn + Bg - 1) / Bg);
    for (int f = threadIdx.x; f < Fpad; f += blockDim.x) {
        float m = 0.0f;
        if (f < F) {
            m = -__builtin_inff();
            for (int n = start; n < end; ++n)
                m = fmaxf(m, (float)x[(size_t)n * ldin + f]);
        }
        pooled[(size_t)g * Fpad + f] = (bf16)m;
    }
}

// ccatp -> xc[:, off..off+Cc) bf16; zero xc[:, off+Cc..off+Cc+pad)
__global__ void copy_ccatp_kernel(const float* __restrict__ cc,
                                  bf16* __restrict__ xc,
                                  int Bg, int Cc, int pad, int ldxc, int off)
{
    int Ct = Cc + pad;
    size_t total = (size_t)Bg * Ct;
    size_t i = (size_t)blockIdx.x * blockDim.x + threadIdx.x;
    size_t st = (size_t)gridDim.x * blockDim.x;
    for (; i < total; i += st) {
        int r = (int)(i / Ct), c = (int)(i % Ct);
        bf16 v = (c < Cc) ? (bf16)cc[(size_t)r * Cc + c] : (bf16)0.0f;
        xc[(size_t)r * ldxc + off + c] = v;
    }
}

// batchnorm over rows (batch dim), one block per column, fused stats+apply
__global__ __launch_bounds__(256)
void bn_kernel(const float* __restrict__ y,
               const float* __restrict__ g, const float* __restrict__ b,
               bf16* __restrict__ out, int Rows, int Cols)
{
    __shared__ float s1[256], s2[256];
    int c = blockIdx.x;
    int tid = threadIdx.x;
    float s = 0.0f, q = 0.0f;
    for (int r = tid; r < Rows; r += 256) {
        float v = y[(size_t)r * Cols + c];
        s += v; q += v * v;
    }
    s1[tid] = s; s2[tid] = q;
    __syncthreads();
    for (int o = 128; o > 0; o >>= 1) {
        if (tid < o) { s1[tid] += s1[tid + o]; s2[tid] += s2[tid + o]; }
        __syncthreads();
    }
    float mean = s1[0] / (float)Rows;
    float var  = s2[0] / (float)Rows - mean * mean;
    float inv  = rsqrtf(var + 1e-5f) * g[c];
    float bb   = b[c];
    for (int r = tid; r < Rows; r += 256)
        out[(size_t)r * Cols + c] = (bf16)((y[(size_t)r * Cols + c] - mean) * inv + bb);
}

// ---------------- host-side orchestration ----------------------------------
static void launch_gemm(const bf16* A, int lda, const bf16* B, int ldb,
                        const float* bias, float* Cf, bf16* Cb, int ldc,
                        int M, int N, int K, int relu, int nstore,
                        hipStream_t s)
{
    dim3 grid((M + BM - 1) / BM, (nstore + BN - 1) / BN);
    gemm_bf16_kernel<<<grid, dim3(256), 0, s>>>(A, lda, B, ldb, bias,
                                                Cf, Cb, ldc, M, N, K, relu,
                                                nstore);
}

static void launch_cvt_rows(const float* in, bf16* out, int rk, int C,
                            int rkpad, hipStream_t s)
{
    size_t n = (size_t)rkpad * C;
    int blocks = (int)((n + 255) / 256);
    if (blocks > 65535) blocks = 65535;
    cvt_pad_rows_kernel<<<blocks, 256, 0, s>>>(in, out, rk, C, rkpad);
}

extern "C" void kernel_launch(void* const* d_in, const int* in_sizes, int n_in,
                              void* d_out, int out_size, void* d_ws, size_t ws_size,
                              hipStream_t stream)
{
    const float* x1  = (const float*)d_in[0];
    const int*   ei1 = (const int*)d_in[1];
    const float* x2  = (const float*)d_in[3];
    const int*   ei2 = (const int*)d_in[4];
    const float* cc  = (const float*)d_in[6];
    const float* P[24];
    for (int i = 0; i < 24; ++i) P[i] = (const float*)d_in[7 + i];
    const float *c1w=P[0],*c1b=P[1],*c2w=P[2],*c2b=P[3],*c3w=P[4],*c3b=P[5],
                *g1w=P[6],*g1b=P[7],*g2w=P[8],*g2b=P[9],
                *fc1w=P[10],*fc1b=P[11],*bn1g=P[12],*bn1b=P[13],
                *fc11w=P[14],*fc11b=P[15],*bn11g=P[16],*bn11b=P[17],
                *fc2w=P[18],*fc2b=P[19],*bn2g=P[20],*bn2b=P[21],
                *outw=P[22],*outb=P[23];

    // workspace carve (256B aligned blocks; weights get over-read slack)
    char* w = (char*)d_ws;
    auto alloc = [&](size_t bytes) -> void* {
        void* p = (void*)w;
        w += (bytes + 255) & ~(size_t)255;
        return p;
    };
    const size_t SLACK = 512;   // safe B-tile over-read past ragged N
    bf16* wc1  = (bf16*)alloc((size_t)80 * 78 * 2 + SLACK);
    bf16* wc2  = (bf16*)alloc((size_t)80 * 156 * 2 + SLACK);
    bf16* wc3  = (bf16*)alloc((size_t)160 * 312 * 2 + SLACK);
    bf16* wg1  = (bf16*)alloc((size_t)320 * 156 * 2 + SLACK);
    bf16* wg2  = (bf16*)alloc((size_t)160 * 128 * 2 + SLACK);
    bf16* wfc1 = (bf16*)alloc((size_t)5760 * 2048 * 2 + SLACK);
    bf16* wfc11= (bf16*)alloc((size_t)2048 * 512 * 2 + SLACK);
    bf16* wfc2 = (bf16*)alloc((size_t)512 * 256 * 2 + SLACK);
    bf16* wout = (bf16*)alloc((size_t)256 * 65 * 2 + SLACK);
    float* dis = (float*)alloc((size_t)GNODES * 4);
    bf16* xb0  = (bf16*)alloc((size_t)GNODES * 320 * 2);
    bf16* xb1  = (bf16*)alloc((size_t)GNODES * 320 * 2);
    bf16* hbuf = (bf16*)alloc((size_t)GNODES * 312 * 2);
    float* obuf= (float*)alloc((size_t)GNODES * 312 * 4);
    bf16* pooled=(bf16*)alloc((size_t)GBATCH * 320 * 2);
    bf16* t1   = (bf16*)alloc((size_t)GBATCH * 160 * 2);
    bf16* xc   = (bf16*)alloc((size_t)GBATCH * 5760 * 2);
    float* y1  = (float*)alloc((size_t)GBATCH * 2048 * 4);
    bf16* y1b  = (bf16*)alloc((size_t)GBATCH * 2048 * 2);
    float* y2  = (float*)alloc((size_t)GBATCH * 512 * 4);
    bf16* y2b  = (bf16*)alloc((size_t)GBATCH * 512 * 2);
    float* y3  = (float*)alloc((size_t)GBATCH * 256 * 4);
    bf16* y3b  = (bf16*)alloc((size_t)GBATCH * 256 * 2);

    // weights fp32 -> bf16 with zero K-pad rows (stateless; reruns each call)
    launch_cvt_rows(c1w,  wc1,  78,   78,   80,   stream);
    launch_cvt_rows(c2w,  wc2,  78,   156,  80,   stream);
    launch_cvt_rows(c3w,  wc3,  156,  312,  160,  stream);
    launch_cvt_rows(g1w,  wg1,  312,  156,  320,  stream);
    launch_cvt_rows(g2w,  wg2,  156,  128,  160,  stream);
    launch_cvt_rows(fc1w, wfc1, 5736, 2048, 5760, stream);
    launch_cvt_rows(fc11w,wfc11,2048, 512,  2048, stream);
    launch_cvt_rows(fc2w, wfc2, 512,  256,  512,  stream);
    launch_cvt_rows(outw, wout, 256,  65,   256,  stream);

    const int convF [4] = {78, 78, 156, 312};    // logical feature widths
    const int convFP[4] = {80, 80, 160, 320};    // padded (ld / K) widths
    const bf16* convW[3] = {wc1, wc2, wc3};
    const float* convB[3] = {c1b, c2b, c3b};

    for (int br = 0; br < 2; ++br) {
        const float* x  = br ? x2 : x1;
        const int*   ei = br ? ei2 : ei1;
        const int* src = ei;
        const int* dst = ei + GEDGES;

        // normalized degrees (with self loop)
        zero_f32_kernel<<<1024, 256, 0, stream>>>(dis, (size_t)GNODES);
        degree_kernel<<<(GEDGES + 255) / 256, 256, 0, stream>>>(dst, dis, GEDGES);
        dis_kernel<<<(GNODES + 255) / 256, 256, 0, stream>>>(dis, GNODES);

        // node features -> padded bf16 [N,80]
        cvt_pad_cols_kernel<<<16384, 256, 0, stream>>>(x, xb0, GNODES, NF0, 80);

        bf16* xin = xb0;
        bf16* xout = xb1;
        for (int l = 0; l < 3; ++l) {
            int Fout = convF[l + 1];
            int Kp   = convFP[l];        // padded input width == GEMM K
            int FoutP= convFP[l + 1];
            // h = x @ W  (bf16 WMMA, f32 accum, stored bf16)
            launch_gemm(xin, Kp, convW[l], Fout, nullptr,
                        nullptr, hbuf, Fout, GNODES, Fout, Kp, 0, Fout, stream);
            // out = h*dis^2 ; out[dst] += h[src]*dis[src]*dis[dst]
            selfloop_kernel<<<4096, 256, 0, stream>>>(hbuf, dis, obuf, GNODES, Fout);
            scatter_kernel<<<GEDGES, 128, 0, stream>>>(hbuf, src, dst, dis,
                                                       obuf, GEDGES, Fout);
            // x_next = relu(out + b)  (bf16, zero-padded cols)
            bias_relu_pad_kernel<<<8192, 256, 0, stream>>>(obuf, convB[l], xout,
                                                           GNODES, Fout, FoutP);
            bf16* t = xin; xin = xout; xout = t;
        }

        // global max pool -> [B,320] (cols 312..319 zero)
        pool_max_kernel<<<GBATCH, 128, 0, stream>>>(xin, pooled, GNODES,
                                                    312, 320, 320, GBATCH);
        // g1: relu(pooled @ g1_w + b) -> t1 [B,160] (cols 156..159 zero)
        launch_gemm(pooled, 320, wg1, 156, g1b, nullptr, t1, 160,
                    GBATCH, 156, 320, 1, 160, stream);
        // g2: t1 @ g2_w + b -> columns [br*128, br*128+128) of xc
        launch_gemm(t1, 160, wg2, 128, g2b, nullptr, xc + br * 128, 5760,
                    GBATCH, 128, 160, 0, 128, stream);
    }

    // ccatp -> xc cols 256..5735; zero xc cols 5736..5759
    copy_ccatp_kernel<<<16384, 256, 0, stream>>>(cc, xc, GBATCH, NCCATP, 24,
                                                 5760, 256);

    // fc1 -> bn1
    launch_gemm(xc, 5760, wfc1, 2048, fc1b, y1, nullptr, 2048,
                GBATCH, 2048, 5760, 1, 2048, stream);
    bn_kernel<<<2048, 256, 0, stream>>>(y1, bn1g, bn1b, y1b, GBATCH, 2048);
    // fc11 -> bn11
    launch_gemm(y1b, 2048, wfc11, 512, fc11b, y2, nullptr, 512,
                GBATCH, 512, 2048, 1, 512, stream);
    bn_kernel<<<512, 256, 0, stream>>>(y2, bn11g, bn11b, y2b, GBATCH, 512);
    // fc2 -> bn2
    launch_gemm(y2b, 512, wfc2, 256, fc2b, y3, nullptr, 256,
                GBATCH, 256, 512, 1, 256, stream);
    bn_kernel<<<256, 256, 0, stream>>>(y3, bn2g, bn2b, y3b, GBATCH, 256);
    // out layer -> d_out [4096,65] fp32
    launch_gemm(y3b, 256, wout, 65, outb, (float*)d_out, nullptr, 65,
                GBATCH, 65, 256, 0, 65, stream);

    (void)in_sizes; (void)n_in; (void)out_size; (void)ws_size;
}
